// PolicyNetRSNNPB_snntorch_74972949119032
// MI455X (gfx1250) — matrix operand order repaired and checked by reading, not compiled, not run on hardware
//
#include <hip/hip_runtime.h>
#include <hip/hip_bf16.h>

// ---------------------------------------------------------------------------
// PolicyNet RSNN (snntorch-style) forward rollout for MI455X / gfx1250.
// Single persistent 32-wave workgroup; bf16 WMMA (v_wmma_f32_16x16x32_bf16),
// spikes live in LDS, f32 neuron state in L2-resident workspace.
// ---------------------------------------------------------------------------

typedef __bf16 bf16_t;
typedef __attribute__((ext_vector_type(8)))  __bf16 v8bf;
typedef __attribute__((ext_vector_type(16))) __bf16 v16bf;
typedef __attribute__((ext_vector_type(8)))  float  v8f;

#define T_STEPS 256
#define NB      128   // batch
#define HID     512
#define INDIM   128   // state 64 + target 64
#define APDIM   64

// ---- workspace byte offsets --------------------------------------------------
#define WRI_OFF  0u
#define WR_OFF   (WRI_OFF + 512u*128u*2u)        // bf16 W_rec_in [512][128]
#define WF_OFF   (WR_OFF  + 512u*512u*2u)        // bf16 W_rec    [512][512]
#define WMU_OFF  (WF_OFF  + 512u*512u*2u)        // bf16 W_ff_in  [512][512]
#define WLV_OFF  (WMU_OFF + 64u*512u*2u)         // bf16 W_mu_in  [64][512]
#define SYNR_OFF (WLV_OFF + 64u*512u*2u)         // bf16 W_lv_in  [64][512]
#define MEMR_OFF (SYNR_OFF + 128u*512u*4u)       // f32 states...
#define SYNF_OFF (MEMR_OFF + 128u*512u*4u)
#define MEMF_OFF (SYNF_OFF + 128u*512u*4u)
#define MMU_OFF  (MEMF_OFF + 128u*512u*4u)
#define MLV_OFF  (MMU_OFF  + 128u*64u*4u)
#define STATE_F32_COUNT (4u*128u*512u + 2u*128u*64u)

// ---- LDS layout (dynamic): 289 KB < 320 KB WGP LDS ---------------------------
// sX  : bf16 [128][128]  x_t                       32768 B
// sSr : bf16 [128][512]  recurrent spikes         131072 B
// sSf : bf16 [128][512]  feed-forward spikes      131072 B
// sMu : f32  [128] row-sum accum                     512 B
// sLv : f32  [128]                                   512 B
#define SMEM_BYTES (32768u + 131072u + 131072u + 512u + 512u)

__device__ __forceinline__ v16bf make16(v8bf lo, v8bf hi) {
    union { v8bf h[2]; v16bf v; } u;
    u.h[0] = lo; u.h[1] = hi;
    return u.v;
}

// A fragment (16x32 bf16, M x K) from LDS, row-major activations.
// Lane m=L&15 holds row M; half=L>>4 selects K sub-range;
// elements 0..7 -> K = k0+8*half.., elements 8..15 -> K = k0+16+8*half..
__device__ __forceinline__ v16bf fragA_lds(const bf16_t* base, int ld,
                                           int row0, int k0, int m, int half) {
    const bf16_t* p = base + (row0 + m) * ld + k0 + 8 * half;
    return make16(*(const v8bf*)p, *(const v8bf*)(p + 16));
}

// B fragment (32x16 bf16, K x N) where B = W^T and W is [H][K] row-major in
// global memory: lane holds column N = (h0+m) -> W row; elements 0..7 -> K = k0+8*half..,
// elements 8..15 -> K = k0+16+8*half.. (same K order as fragA_lds).
__device__ __forceinline__ v16bf fragB_glb(const bf16_t* base, int ld,
                                           int h0, int k0, int m, int half) {
    const bf16_t* p = base + (h0 + m) * ld + k0 + 8 * half;
    return make16(*(const v8bf*)p, *(const v8bf*)(p + 16));
}

// Warm the WGP-level cache for the next B fragment. Scope 0 (WGP) is the
// encoding default (no mnemonic token); ISA: WGP-scope prefetch pulls the
// line into ALL cache levels on miss (SE/DEV scopes only reach GL2, which
// the weights already live in). Non-speculative is fine: every prefetched
// address is inside the mapped workspace.
__device__ __forceinline__ void prefB_glb(const bf16_t* base, int ld,
                                          int h0, int k0, int m, int half) {
    const bf16_t* p = base + (h0 + m) * ld + k0 + 16 * half;
    asm volatile("global_prefetch_b8 %0, off" : : "v"(p) : "memory");
}

#define WMMA_BF16(a, b, c) \
    __builtin_amdgcn_wmma_f32_16x16x32_bf16(false, (a), false, (b), (short)0, (c), false, false)

// ---------------------------------------------------------------------------
// Prep: weights fp32 -> bf16 in workspace; zero all neuron state (every call,
// so graph replays are deterministic).
// ---------------------------------------------------------------------------
__global__ void rsnn_prep_kernel(const float* __restrict__ Wri,
                                 const float* __restrict__ Wr,
                                 const float* __restrict__ Wf,
                                 const float* __restrict__ Wmu,
                                 const float* __restrict__ Wlv,
                                 unsigned char* __restrict__ ws) {
    bf16_t* oWri = (bf16_t*)(ws + WRI_OFF);
    bf16_t* oWr  = (bf16_t*)(ws + WR_OFF);
    bf16_t* oWf  = (bf16_t*)(ws + WF_OFF);
    bf16_t* oWmu = (bf16_t*)(ws + WMU_OFF);
    bf16_t* oWlv = (bf16_t*)(ws + WLV_OFF);
    float*  st   = (float*)(ws + SYNR_OFF);

    const int tid = blockIdx.x * blockDim.x + threadIdx.x;
    const int np  = gridDim.x * blockDim.x;

    for (int i = tid; i < 512 * 128; i += np) oWri[i] = (bf16_t)Wri[i];
    for (int i = tid; i < 512 * 512; i += np) oWr[i]  = (bf16_t)Wr[i];
    for (int i = tid; i < 512 * 512; i += np) oWf[i]  = (bf16_t)Wf[i];
    for (int i = tid; i < 64 * 512;  i += np) oWmu[i] = (bf16_t)Wmu[i];
    for (int i = tid; i < 64 * 512;  i += np) oWlv[i] = (bf16_t)Wlv[i];
    for (int i = tid; i < (int)STATE_F32_COUNT; i += np) st[i] = 0.0f;
}

// ---------------------------------------------------------------------------
// Persistent rollout kernel: 1 block, 1024 threads = 32 waves.
// Wave w owns HID column tile [16w, 16w+16) for the two big GEMMs, and
// (mu,lv) tile (row = w>>2, col = w&3) for the readout GEMMs.
// ---------------------------------------------------------------------------
__global__ void __launch_bounds__(1024, 1)
rsnn_rollout_kernel(const float* __restrict__ state,
                    const float* __restrict__ target,
                    const float* __restrict__ b_rec_in,
                    const float* __restrict__ b_rec,
                    const float* __restrict__ alpha_rec,
                    const float* __restrict__ beta_rec,
                    const float* __restrict__ b_ff_in,
                    const float* __restrict__ alpha_ff,
                    const float* __restrict__ beta_ff,
                    const float* __restrict__ b_mu_in,
                    const float* __restrict__ beta_mu,
                    const float* __restrict__ b_lv_in,
                    const float* __restrict__ beta_lv,
                    unsigned char* __restrict__ ws,
                    float* __restrict__ out) {
    extern __shared__ unsigned char smem[];
    bf16_t* sX  = (bf16_t*)smem;
    bf16_t* sSr = (bf16_t*)(smem + 32768u);
    bf16_t* sSf = (bf16_t*)(smem + 32768u + 131072u);
    float*  sMu = (float*)(smem + 32768u + 2u * 131072u);
    float*  sLv = (float*)(smem + 32768u + 2u * 131072u + 512u);

    const bf16_t* Wri = (const bf16_t*)(ws + WRI_OFF);
    const bf16_t* Wr  = (const bf16_t*)(ws + WR_OFF);
    const bf16_t* Wf  = (const bf16_t*)(ws + WF_OFF);
    const bf16_t* Wmu = (const bf16_t*)(ws + WMU_OFF);
    const bf16_t* Wlv = (const bf16_t*)(ws + WLV_OFF);
    float* wsSynR = (float*)(ws + SYNR_OFF);
    float* wsMemR = (float*)(ws + MEMR_OFF);
    float* wsSynF = (float*)(ws + SYNF_OFF);
    float* wsMemF = (float*)(ws + MEMF_OFF);
    float* wsMMu  = (float*)(ws + MMU_OFF);
    float* wsMLv  = (float*)(ws + MLV_OFF);

    const int tid  = threadIdx.x;
    const int m    = tid & 15;         // N index inside tile / W row within tile
    const int half = (tid >> 4) & 1;   // lane half
    const int w    = tid >> 5;         // wave id 0..31

    // Per-lane constant neuron params (fixed h / ap per lane for whole kernel).
    const int h  = (w << 4) + m;       // HID column for phases A/B
    const int cc = w & 3;              // readout col tile
    const int rc = w >> 2;             // readout row tile
    const int ap = (cc << 4) + m;      // AP column for phase C

    const float aR    = alpha_rec[h];
    const float bR    = beta_rec[h];
    const float biasR = b_rec_in[h] + b_rec[h];
    const float aF    = alpha_ff[h];
    const float bF    = beta_ff[h];
    const float biasF = b_ff_in[h];
    const float bMu   = beta_mu[ap];
    const float biasM = b_mu_in[ap];
    const float bLv   = beta_lv[ap];
    const float biasL = b_lv_in[ap];

    // Initial recurrent spikes = 0.
    for (int i = tid; i < NB * HID; i += 1024) sSr[i] = (bf16_t)0.0f;
    __syncthreads();

    for (int t = 0; t < T_STEPS; ++t) {
        // Zero mu/lv row-sum accumulators; stage x_t = [state|target] as bf16.
        if (tid < NB) { sMu[tid] = 0.0f; sLv[tid] = 0.0f; }
        for (int i = tid; i < NB * INDIM; i += 1024) {
            const int n = i >> 7, k = i & 127;
            const float v = (k < 64) ? state[(size_t)(t * NB + n) * 64 + k]
                                     : target[(size_t)(t * NB + n) * 64 + (k - 64)];
            sX[i] = (bf16_t)v;
        }
        __syncthreads();

#pragma unroll 1
        for (int rep = 0; rep < 2; ++rep) {
            // ---------------- Phase A: cur_r = x@Wri^T + spk_r@Wr^T ----------
            v8f acc[8];
#pragma unroll
            for (int r = 0; r < 8; ++r) acc[r] = (v8f)0.0f;

            prefB_glb(Wri, INDIM, w << 4, 0, m, half);
#pragma unroll 1
            for (int k = 0; k < INDIM; k += 32) {
                const v16bf bfr = fragB_glb(Wri, INDIM, w << 4, k, m, half);
                // Warm next B chunk (last iter prefetches into Wr's slab; harmless).
                prefB_glb(Wri, INDIM, w << 4, k + 32, m, half);
#pragma unroll
                for (int r = 0; r < 8; ++r) {
                    const v16bf afr = fragA_lds(sX, INDIM, r << 4, k, m, half);
                    acc[r] = WMMA_BF16(afr, bfr, acc[r]);
                }
            }
            prefB_glb(Wr, HID, w << 4, 0, m, half);
#pragma unroll 1
            for (int k = 0; k < HID; k += 32) {
                const v16bf bfr = fragB_glb(Wr, HID, w << 4, k, m, half);
                prefB_glb(Wr, HID, w << 4, k + 32, m, half);
#pragma unroll
                for (int r = 0; r < 8; ++r) {
                    const v16bf afr = fragA_lds(sSr, HID, r << 4, k, m, half);
                    acc[r] = WMMA_BF16(afr, bfr, acc[r]);
                }
            }
            __syncthreads();  // all waves done reading old sSr

            // LIF update in WMMA C-layout: lane's (n,h) = (16r+8*half+v, h).
#pragma unroll
            for (int r = 0; r < 8; ++r) {
#pragma unroll
                for (int v = 0; v < 8; ++v) {
                    const int n   = (r << 4) + (half << 3) + v;
                    const int idx = n * HID + h;
                    const float cur  = acc[r][v] + biasR;
                    float syn  = wsSynR[idx];
                    float memv = wsMemR[idx];
                    const float reset = (memv > 1.0f) ? 1.0f : 0.0f;
                    syn  = aR * syn + cur;
                    memv = bR * memv + syn - reset;   // THR = 1.0
                    wsSynR[idx] = syn;
                    wsMemR[idx] = memv;
                    sSr[idx] = (bf16_t)((memv > 1.0f) ? 1.0f : 0.0f);
                }
            }
            __syncthreads();  // new spk_r visible

            // ---------------- Phase B: cur_f = spk_r@Wf^T --------------------
#pragma unroll
            for (int r = 0; r < 8; ++r) acc[r] = (v8f)0.0f;
            prefB_glb(Wf, HID, w << 4, 0, m, half);
#pragma unroll 1
            for (int k = 0; k < HID; k += 32) {
                const v16bf bfr = fragB_glb(Wf, HID, w << 4, k, m, half);
                prefB_glb(Wf, HID, w << 4, k + 32, m, half);
#pragma unroll
                for (int r = 0; r < 8; ++r) {
                    const v16bf afr = fragA_lds(sSr, HID, r << 4, k, m, half);
                    acc[r] = WMMA_BF16(afr, bfr, acc[r]);
                }
            }
            __syncthreads();

#pragma unroll
            for (int r = 0; r < 8; ++r) {
#pragma unroll
                for (int v = 0; v < 8; ++v) {
                    const int n   = (r << 4) + (half << 3) + v;
                    const int idx = n * HID + h;
                    const float cur  = acc[r][v] + biasF;
                    float syn  = wsSynF[idx];
                    float memv = wsMemF[idx];
                    const float reset = (memv > 1.0f) ? 1.0f : 0.0f;
                    syn  = aF * syn + cur;
                    memv = bF * memv + syn - reset;
                    wsSynF[idx] = syn;
                    wsMemF[idx] = memv;
                    sSf[idx] = (bf16_t)((memv > 1.0f) ? 1.0f : 0.0f);
                }
            }
            __syncthreads();  // spk_f visible

            // ---------------- Phase C: readout mu/lv -------------------------
            v8f amu = (v8f)0.0f, alv = (v8f)0.0f;
            prefB_glb(Wmu, HID, cc << 4, 0, m, half);
            prefB_glb(Wlv, HID, cc << 4, 0, m, half);
#pragma unroll 1
            for (int k = 0; k < HID; k += 32) {
                const v16bf afr  = fragA_lds(sSf, HID, rc << 4, k, m, half);
                const v16bf bmu  = fragB_glb(Wmu, HID, cc << 4, k, m, half);
                const v16bf blv  = fragB_glb(Wlv, HID, cc << 4, k, m, half);
                prefB_glb(Wmu, HID, cc << 4, k + 32, m, half);
                prefB_glb(Wlv, HID, cc << 4, k + 32, m, half);
                amu = WMMA_BF16(afr, bmu, amu);
                alv = WMMA_BF16(afr, blv, alv);
            }
            // mem update + row-sum (W_mu_out == full(1/P) -> mu = rowsum/8).
#pragma unroll
            for (int v = 0; v < 8; ++v) {
                const int n   = (rc << 4) + (half << 3) + v;
                const int idx = n * APDIM + ap;
                float mm = wsMMu[idx];
                mm = bMu * mm + amu[v] + biasM;
                wsMMu[idx] = mm;
                atomicAdd(&sMu[n], mm);
                float ml = wsMLv[idx];
                ml = bLv * ml + alv[v] + biasL;
                wsMLv[idx] = ml;
                atomicAdd(&sLv[n], ml);
            }
            __syncthreads();
        }

        // mu[t,n,a] = (rowsum_step1 + rowsum_step2) * (1/P) / REPEAT = sum/16.
        {
            const int n = tid >> 3;  // tid in [0,1024): n=tid/8, a=tid%8
            out[(size_t)t * (NB * 8) + tid]                            = sMu[n] * 0.0625f;
            out[(size_t)T_STEPS * NB * 8 + (size_t)t * (NB * 8) + tid] = sLv[n] * 0.0625f;
        }
        __syncthreads();  // before next t overwrites sX / sMu / sLv
    }
}

extern "C" void kernel_launch(void* const* d_in, const int* in_sizes, int n_in,
                              void* d_out, int out_size, void* d_ws, size_t ws_size,
                              hipStream_t stream) {
    (void)in_sizes; (void)n_in; (void)out_size; (void)ws_size;

    const float* state     = (const float*)d_in[0];
    const float* target    = (const float*)d_in[1];
    const float* W_rec_in  = (const float*)d_in[2];
    const float* b_rec_in  = (const float*)d_in[3];
    const float* W_rec     = (const float*)d_in[4];
    const float* b_rec     = (const float*)d_in[5];
    const float* alpha_rec = (const float*)d_in[6];
    const float* beta_rec  = (const float*)d_in[7];
    const float* W_ff_in   = (const float*)d_in[8];
    const float* b_ff_in   = (const float*)d_in[9];
    const float* alpha_ff  = (const float*)d_in[10];
    const float* beta_ff   = (const float*)d_in[11];
    const float* W_mu_in   = (const float*)d_in[12];
    const float* b_mu_in   = (const float*)d_in[13];
    const float* beta_mu   = (const float*)d_in[14];
    // d_in[15] = W_mu_out (constant full(1/P), folded into 1/16 scale)
    const float* W_lv_in   = (const float*)d_in[16];
    const float* b_lv_in   = (const float*)d_in[17];
    const float* beta_lv   = (const float*)d_in[18];
    // d_in[19] = W_lv_out (constant full(1/P))

    unsigned char* ws = (unsigned char*)d_ws;
    float* out = (float*)d_out;

    rsnn_prep_kernel<<<dim3(256), dim3(256), 0, stream>>>(
        W_rec_in, W_rec, W_ff_in, W_mu_in, W_lv_in, ws);

    rsnn_rollout_kernel<<<dim3(1), dim3(1024), SMEM_BYTES, stream>>>(
        state, target, b_rec_in, b_rec, alpha_rec, beta_rec,
        b_ff_in, alpha_ff, beta_ff,
        b_mu_in, beta_mu, b_lv_in, beta_lv,
        ws, out);
}
